// NodeEdgeBlock_27960237097249
// MI455X (gfx1250) — matrix operand order repaired and checked by reading, not compiled, not run on hardware
//
#include <hip/hip_runtime.h>
#include <hip/hip_bf16.h>

typedef _Float16 v16h __attribute__((ext_vector_type(16)));
typedef float    v8f  __attribute__((ext_vector_type(8)));

#define BSZ 4
#define NN  256
#define DX  256
#define DE  128
#define DY  128
#define NH  8
#define DF  32
#define RSQRT_DF 0.17677669529663687f

static __device__ inline v8f wmma_f16(v16h a, v16h b, v8f c) {
    // D = A(16x32 f16) * B(32x16 f16) + C(16x16 f32)
    return __builtin_amdgcn_wmma_f32_16x16x32_f16(
        /*neg_a=*/false, a, /*neg_b=*/false, b,
        /*c_mod=*/(short)0, c, /*reuse_a=*/false, /*reuse_b=*/false);
}

union FragU { v16h v; uint4 q[2]; };

// Load A fragment from a row-major [16][K] f16 LDS tile.
// ISA 16-bit A layout: lane (&15) = M row; half h -> K = h + 8*(h>>3) + 8*laneHi,
// i.e. dwords {0..3,8..11} + 4*laneHi of the 16-dword k-slice -> two b128 loads.
static __device__ inline v16h load_a_frag(const _Float16* tile, int rowStrideBytes,
                                          int kt, int lane) {
    int laneHi = (lane >> 4) & 1;
    const char* p = (const char*)tile + (lane & 15) * rowStrideBytes + kt * 64 + laneHi * 16;
    FragU u;
    u.q[0] = *(const uint4*)(p);
    u.q[1] = *(const uint4*)(p + 32);
    return u.v;
}

// Load B fragment from fragment-ready global layout:
// frag[((kt*NT + nt)*32 + lane)*16 + h]
static __device__ inline v16h load_b_frag(const _Float16* frag, int kt, int NT,
                                          int nt, int lane) {
    const uint4* p = (const uint4*)(frag + (((size_t)(kt * NT + nt) * 32 + lane) << 4));
    FragU u;
    u.q[0] = p[0];
    u.q[1] = p[1];
    return u.v;
}

// Async contiguous 64B-per-lane copy global->LDS (4x b128), tracked on ASYNCcnt.
static __device__ inline void async_copy64(unsigned lds_off, const void* gaddr) {
    asm volatile(
        "global_load_async_to_lds_b128 %0, %1, off\n\t"
        "global_load_async_to_lds_b128 %0, %1, off offset:16\n\t"
        "global_load_async_to_lds_b128 %0, %1, off offset:32\n\t"
        "global_load_async_to_lds_b128 %0, %1, off offset:48"
        :: "v"(lds_off), "v"(gaddr) : "memory");
}

static __device__ inline void wait_async0() {
    asm volatile("s_wait_asynccnt 0x0" ::: "memory");
}

// ---------------------------------------------------------------------------
// Pack a K x Nn f32 weight into fragment-ready f16 B layout.
// B layout (ISA 7.12.5): lane -> N = lane&15; K = kt*32 + (h&7) + 8*laneHi + 16*(h>>3).
// ---------------------------------------------------------------------------
__global__ void pack_b_frag_kernel(const float* __restrict__ W, _Float16* __restrict__ dst,
                                   int K, int Nn) {
    int idx = blockIdx.x * blockDim.x + threadIdx.x;
    if (idx >= K * Nn) return;
    int h    = idx & 15;
    int lane = (idx >> 4) & 31;
    int fr   = idx >> 9;
    int NT   = Nn >> 4;
    int nt   = fr % NT;
    int kt   = fr / NT;
    int hi   = (lane >> 4) & 1;
    int Kidx = kt * 32 + (h & 7) + 8 * hi + 16 * (h >> 3);
    int Nidx = nt * 16 + (lane & 15);
    dst[idx] = (_Float16)W[Kidx * Nn + Nidx];
}

// ---------------------------------------------------------------------------
// q,k,v = (x @ W{q,k,v}) * x_mask   (one block per (b,i))
// ---------------------------------------------------------------------------
__global__ __launch_bounds__(256) void qkv_kernel(
    const float* __restrict__ x, const int* __restrict__ node_mask,
    const float* __restrict__ Wq, const float* __restrict__ Wk, const float* __restrict__ Wv,
    float* __restrict__ qbuf, float* __restrict__ kbuf, float* __restrict__ vbuf) {
    __shared__ float sx[DX];
    int row = blockIdx.x;           // b*N + i
    int tid = threadIdx.x;
    sx[tid] = x[(size_t)row * DX + tid];
    __syncthreads();
    float m = (node_mask[row] > 0) ? 1.f : 0.f;
    float aq = 0.f, ak = 0.f, av = 0.f;
    #pragma unroll 4
    for (int c = 0; c < DX; ++c) {
        float xv = sx[c];
        aq += xv * Wq[c * DX + tid];
        ak += xv * Wk[c * DX + tid];
        av += xv * Wv[c * DX + tid];
    }
    size_t o = (size_t)row * DX + tid;
    qbuf[o] = aq * m;
    kbuf[o] = ak * m;
    vbuf[o] = av * m;
}

// ---------------------------------------------------------------------------
// Per-batch y-derived vectors, x mean over nodes, zero e-sum accumulator.
// ---------------------------------------------------------------------------
__global__ __launch_bounds__(256) void yvec_kernel(
    const float* __restrict__ x, const float* __restrict__ y,
    const float* __restrict__ Wye_add, const float* __restrict__ Wye_mul,
    const float* __restrict__ Wyx_add, const float* __restrict__ Wyx_mul,
    const float* __restrict__ Wyy,
    float* __restrict__ yeA, float* __restrict__ yeM,
    float* __restrict__ yxA, float* __restrict__ yxM,
    float* __restrict__ ylin, float* __restrict__ xmean, float* __restrict__ esum_g) {
    __shared__ float sy[DY];
    int b = blockIdx.x;
    int tid = threadIdx.x;
    if (tid < DY) sy[tid] = y[b * DY + tid];
    __syncthreads();
    float a1 = 0.f, a2 = 0.f, a3 = 0.f, a4 = 0.f;
    #pragma unroll 4
    for (int c = 0; c < DY; ++c) {
        float yv = sy[c];
        a1 += yv * Wye_add[c * DX + tid];
        a2 += yv * Wye_mul[c * DX + tid];
        a3 += yv * Wyx_add[c * DX + tid];
        a4 += yv * Wyx_mul[c * DX + tid];
    }
    yeA[b * DX + tid] = a1;
    yeM[b * DX + tid] = a2;
    yxA[b * DX + tid] = a3;
    yxM[b * DX + tid] = a4;
    if (tid < DY) {
        float a = 0.f;
        for (int c = 0; c < DY; ++c) a += sy[c] * Wyy[c * DY + tid];
        ylin[b * DY + tid] = a;
    }
    if (tid < DE) esum_g[b * DE + tid] = 0.f;
    float s = 0.f;
    for (int i = 0; i < NN; ++i) s += x[((size_t)b * NN + i) * DX + tid];
    xmean[b * DX + tid] = s * (1.f / (float)NN);
}

// ---------------------------------------------------------------------------
// Fused main kernel: one workgroup (8 wave32) per (b, i).
// Sweeps j in 16-wide tiles:  E-tile -> WMMA(e1,e2) -> attn/FiLM/online-softmax
// -> P-tile -> WMMA(We_out) -> new_e.   Then merged softmax -> new_x row.
// k/v tiles staged with GLOBAL_LOAD_ASYNC_TO_LDS_B128 (ASYNCcnt); next e tile
// prefetched with global_prefetch_b8 behind the current tile's WMMAs.
// ---------------------------------------------------------------------------
__global__ __launch_bounds__(256) void fused_main_kernel(
    const float* __restrict__ e, const int* __restrict__ node_mask,
    const float* __restrict__ qbuf, const float* __restrict__ kbuf,
    const float* __restrict__ vbuf,
    const float* __restrict__ yeA, const float* __restrict__ yeM,
    const float* __restrict__ yxA, const float* __restrict__ yxM,
    const _Float16* __restrict__ fragMul, const _Float16* __restrict__ fragAdd,
    const _Float16* __restrict__ fragOut, const float* __restrict__ Wx_out,
    float* __restrict__ out_newx, float* __restrict__ out_newe,
    float* __restrict__ esum_g) {
    __shared__ __align__(16) _Float16 sE[16 * DE];   // masked e tile (A for GEMM1)
    __shared__ __align__(16) _Float16 sP[16 * DX];   // FiLM'd logits (A for GEMM2)
    __shared__ __align__(16) float sK[16 * DX];
    __shared__ __align__(16) float sV[16 * DX];
    __shared__ float sQ[DX];
    __shared__ float sYeA[DX], sYeM[DX], sYxA[DX], sYxM[DX];
    __shared__ float sMask[NN];
    __shared__ float sW[DX];
    __shared__ float sEsum[DE];

    const int tid    = threadIdx.x;
    const int wave   = tid >> 5;
    const int lane   = tid & 31;
    const int laneHi = (lane >> 4) & 1;
    const int ln     = lane & 15;
    const int b      = blockIdx.x / NN;
    const int i      = blockIdx.x % NN;

    const float mask_i = (node_mask[b * NN + i] > 0) ? 1.f : 0.f;

    sQ[tid]    = qbuf[((size_t)(b * NN + i)) * DX + tid];
    sYeA[tid]  = yeA[b * DX + tid];
    sYeM[tid]  = yeM[b * DX + tid];
    sYxA[tid]  = yxA[b * DX + tid];
    sYxM[tid]  = yxM[b * DX + tid];
    sMask[tid] = (node_mask[b * NN + tid] > 0) ? 1.f : 0.f;
    if (tid < DE) sEsum[tid] = 0.f;
    __syncthreads();

    const int nt0 = 2 * wave;                     // GEMM1 N-tiles: nt0, nt0+1
    float qr[2], yeAr[2], yeMr[2];
    #pragma unroll
    for (int l2 = 0; l2 < 2; ++l2) {
        int c = (nt0 + l2) * 16 + ln;
        qr[l2]   = sQ[c] * RSQRT_DF;              // pre-scale q by 1/sqrt(df)
        yeAr[l2] = sYeA[c];
        yeMr[l2] = sYeM[c];
    }
    float sm_m[2] = {-1e30f, -1e30f};
    float sm_l[2] = {0.f, 0.f};
    float sm_a[2] = {0.f, 0.f};

    const int r_row = tid >> 4;                   // e staging row  (0..15)
    const int c8    = (tid & 15) * 8;             // e channels     (0..120)
    float es[8] = {0.f, 0.f, 0.f, 0.f, 0.f, 0.f, 0.f, 0.f};

    // per-thread LDS byte offsets / global base for async k/v tile copies
    const unsigned ldsK = (unsigned)(size_t)(void*)&sK[tid * 16];
    const unsigned ldsV = (unsigned)(size_t)(void*)&sV[tid * 16];
    const float* ktile_base = kbuf + ((size_t)(b * NN)) * DX + tid * 16;
    const float* vtile_base = vbuf + ((size_t)(b * NN)) * DX + tid * 16;

    for (int jt = 0; jt < 16; ++jt) {
        // ---- async-stage K,V tiles (contiguous 16KB each) ----
        async_copy64(ldsK, ktile_base + (size_t)jt * 16 * DX);
        async_copy64(ldsV, vtile_base + (size_t)jt * 16 * DX);

        // ---- stage E (masked, f16) on VALU path; accumulate raw e-sum ----
        {
            int j = jt * 16 + r_row;
            const float* ep = e + (((size_t)(b * NN + i)) * NN + j) * DE + c8;
            if (jt < 15) __builtin_prefetch(ep + 16 * DE, 0, 0);   // next j-tile
            float mm = mask_i * sMask[j];
            #pragma unroll
            for (int u = 0; u < 8; ++u) {
                float v = ep[u];
                es[u] += v;
                sE[r_row * DE + c8 + u] = (_Float16)(v * mm);
            }
        }
        wait_async0();
        __syncthreads();

        // ---- GEMM1: E(16x128) @ {We_mul, We_add}(128x32 per wave) ----
        v8f acc1[2] = {{0,0,0,0,0,0,0,0}, {0,0,0,0,0,0,0,0}};
        v8f acc2[2] = {{0,0,0,0,0,0,0,0}, {0,0,0,0,0,0,0,0}};
        #pragma unroll
        for (int kt = 0; kt < 4; ++kt) {
            v16h a = load_a_frag(sE, DE * 2, kt, lane);
            #pragma unroll
            for (int l2 = 0; l2 < 2; ++l2) {
                v16h b1 = load_b_frag(fragMul, kt, 16, nt0 + l2, lane);
                acc1[l2] = wmma_f16(a, b1, acc1[l2]);
                v16h b2 = load_b_frag(fragAdd, kt, 16, nt0 + l2, lane);
                acc2[l2] = wmma_f16(a, b2, acc2[l2]);
            }
        }

        // ---- epilogue: attn, branchless per-channel online softmax, FiLM -> P
        #pragma unroll
        for (int l2 = 0; l2 < 2; ++l2) {
            int c = (nt0 + l2) * 16 + ln;
            #pragma unroll
            for (int r = 0; r < 8; ++r) {
                int jl = r + 8 * laneHi;          // C layout: M = r + 8*laneHi
                int j  = jt * 16 + jl;
                float kv   = sK[jl * DX + c];
                float attn = (qr[l2] * kv) * (acc1[l2][r] + 1.f) + acc2[l2][r];
                float s    = (sMask[j] > 0.5f) ? attn : -1e9f;
                float mnew = fmaxf(sm_m[l2], s);
                float sc   = __expf(sm_m[l2] - mnew);
                float w    = __expf(s - mnew);
                sm_m[l2]   = mnew;
                sm_l[l2]   = sm_l[l2] * sc + w;
                sm_a[l2]   = sm_a[l2] * sc + w * sV[jl * DX + c];
                float p = yeAr[l2] + (yeMr[l2] + 1.f) * attn;
                sP[jl * DX + c] = (_Float16)p;
            }
        }
        __syncthreads();

        // ---- GEMM2: P(16x256) @ We_out(256x16 per wave) -> new_e ----
        v8f acco = {0,0,0,0,0,0,0,0};
        #pragma unroll
        for (int kt = 0; kt < 8; ++kt) {
            v16h a  = load_a_frag(sP, DX * 2, kt, lane);
            v16h bw = load_b_frag(fragOut, kt, 8, wave, lane);
            acco = wmma_f16(a, bw, acco);
        }
        int cout = wave * 16 + ln;
        #pragma unroll
        for (int r = 0; r < 8; ++r) {
            int jl = r + 8 * laneHi;
            int j  = jt * 16 + jl;
            float val = acco[r] * mask_i * sMask[j];
            out_newe[(((size_t)(b * NN + i)) * NN + j) * DE + cout] = val;
        }
        __syncthreads();
    }

    // ---- merge lane-half softmax partials, build weighted vector ----
    #pragma unroll
    for (int l2 = 0; l2 < 2; ++l2) {
        float om = __shfl_xor(sm_m[l2], 16, 32);
        float ol = __shfl_xor(sm_l[l2], 16, 32);
        float oa = __shfl_xor(sm_a[l2], 16, 32);
        float M  = fmaxf(sm_m[l2], om);
        float lt = sm_l[l2] * __expf(sm_m[l2] - M) + ol * __expf(om - M);
        float at = sm_a[l2] * __expf(sm_m[l2] - M) + oa * __expf(om - M);
        if (laneHi == 0) {
            int c = (nt0 + l2) * 16 + ln;
            float wgt = at / lt;
            sW[c] = sYxA[c] + (sYxM[c] + 1.f) * wgt;
        }
    }
    #pragma unroll
    for (int u = 0; u < 8; ++u) atomicAdd(&sEsum[c8 + u], es[u]);
    __syncthreads();
    if (tid < DE) atomicAdd(&esum_g[b * DE + tid], sEsum[tid]);

    // ---- new_x row: (yx1 + (yx2+1)*weighted) @ Wx_out * mask_i ----
    float acc = 0.f;
    #pragma unroll 4
    for (int c = 0; c < DX; ++c) acc += sW[c] * Wx_out[c * DX + tid];
    out_newx[((size_t)(b * NN + i)) * DX + tid] = acc * mask_i;
}

// ---------------------------------------------------------------------------
// new_y = relu((y_lin + xmean@Wxy + (esum/N^2)@Wey) @ Wy1) @ Wy2
// ---------------------------------------------------------------------------
__global__ __launch_bounds__(128) void ynew_kernel(
    const float* __restrict__ ylin, const float* __restrict__ xmean,
    const float* __restrict__ esum_g,
    const float* __restrict__ Wxy, const float* __restrict__ Wey,
    const float* __restrict__ Wy1, const float* __restrict__ Wy2,
    float* __restrict__ out_newy) {
    __shared__ float sxm[DX], ses[DE], st[DY], sh[DY];
    int b = blockIdx.x, tid = threadIdx.x;
    sxm[tid]      = xmean[b * DX + tid];
    sxm[tid + DY] = xmean[b * DX + tid + DY];
    ses[tid]      = esum_g[b * DE + tid] * (1.f / (float)(NN * NN));
    __syncthreads();
    float t = ylin[b * DY + tid];
    for (int c = 0; c < DX; ++c) t += sxm[c] * Wxy[c * DY + tid];
    for (int c = 0; c < DE; ++c) t += ses[c] * Wey[c * DY + tid];
    st[tid] = t;
    __syncthreads();
    float h = 0.f;
    for (int c = 0; c < DY; ++c) h += st[c] * Wy1[c * DY + tid];
    sh[tid] = fmaxf(h, 0.f);
    __syncthreads();
    float o = 0.f;
    for (int c = 0; c < DY; ++c) o += sh[c] * Wy2[c * DY + tid];
    out_newy[b * DY + tid] = o;
}

// ---------------------------------------------------------------------------
extern "C" void kernel_launch(void* const* d_in, const int* in_sizes, int n_in,
                              void* d_out, int out_size, void* d_ws, size_t ws_size,
                              hipStream_t stream) {
    const float* x       = (const float*)d_in[0];
    const float* e       = (const float*)d_in[1];
    const float* y       = (const float*)d_in[2];
    const int*   nmask   = (const int*)d_in[3];
    const float* Wq      = (const float*)d_in[4];
    const float* Wk      = (const float*)d_in[5];
    const float* Wv      = (const float*)d_in[6];
    const float* We_mul  = (const float*)d_in[7];
    const float* We_add  = (const float*)d_in[8];
    const float* Wye_mul = (const float*)d_in[9];
    const float* Wye_add = (const float*)d_in[10];
    const float* Wyx_mul = (const float*)d_in[11];
    const float* Wyx_add = (const float*)d_in[12];
    const float* Wyy     = (const float*)d_in[13];
    const float* Wx_out  = (const float*)d_in[14];
    const float* We_out  = (const float*)d_in[15];
    const float* Wxy     = (const float*)d_in[16];
    const float* Wey     = (const float*)d_in[17];
    const float* Wy1     = (const float*)d_in[18];
    const float* Wy2     = (const float*)d_in[19];

    float* out      = (float*)d_out;
    float* out_newx = out;                                   // 4*256*256
    float* out_newe = out + (size_t)BSZ * NN * NN;           // 4*256*256*128
    float* out_newy = out_newe + (size_t)BSZ * NN * NN * DE; // 4*128

    // workspace carve (all 16B aligned)
    float* qbuf  = (float*)d_ws;                 // 262144
    float* kbuf  = qbuf + (size_t)BSZ * NN * DX;
    float* vbuf  = kbuf + (size_t)BSZ * NN * DX;
    float* yeA   = vbuf + (size_t)BSZ * NN * DX; // 1024
    float* yeM   = yeA + BSZ * DX;
    float* yxA   = yeM + BSZ * DX;
    float* yxM   = yxA + BSZ * DX;
    float* ylin  = yxM + BSZ * DX;               // 512
    float* xmean = ylin + BSZ * DY;              // 1024
    float* esum  = xmean + BSZ * DX;             // 512
    _Float16* fragMul = (_Float16*)(esum + BSZ * DE);  // 128*256 halfs
    _Float16* fragAdd = fragMul + DE * DX;
    _Float16* fragOut = fragAdd + DE * DX;             // 256*128 halfs

    // pack weights into fragment-ready f16 B layout
    pack_b_frag_kernel<<<(DE * DX) / 256, 256, 0, stream>>>(We_mul, fragMul, DE, DX);
    pack_b_frag_kernel<<<(DE * DX) / 256, 256, 0, stream>>>(We_add, fragAdd, DE, DX);
    pack_b_frag_kernel<<<(DX * DE) / 256, 256, 0, stream>>>(We_out, fragOut, DX, DE);

    qkv_kernel<<<BSZ * NN, 256, 0, stream>>>(x, nmask, Wq, Wk, Wv, qbuf, kbuf, vbuf);
    yvec_kernel<<<BSZ, 256, 0, stream>>>(x, y, Wye_add, Wye_mul, Wyx_add, Wyx_mul, Wyy,
                                         yeA, yeM, yxA, yxM, ylin, xmean, esum);

    fused_main_kernel<<<BSZ * NN, 256, 0, stream>>>(
        e, nmask, qbuf, kbuf, vbuf, yeA, yeM, yxA, yxM,
        fragMul, fragAdd, fragOut, Wx_out, out_newx, out_newe, esum);

    ynew_kernel<<<BSZ, 128, 0, stream>>>(ylin, xmean, esum, Wxy, Wey, Wy1, Wy2, out_newy);
}
